// Quantize_37512244363889
// MI455X (gfx1250) — matrix operand order repaired and checked by reading, not compiled, not run on hardware
//
#include <hip/hip_runtime.h>
#include <hip/hip_bf16.h>
#include <math.h>

#define DIM   512
#define NEMB  1024
#define NROWS 32768                 // 64*512 rows of x
#define QN    16777216              // 64*512*512 output elements of quantize

typedef __attribute__((ext_vector_type(4)))  _Float16 v4h;
typedef __attribute__((ext_vector_type(8)))  _Float16 v8h;
typedef __attribute__((ext_vector_type(16))) _Float16 v16h;
typedef __attribute__((ext_vector_type(8)))  float    v8f;

union H16 { v16h v; v8h h[2]; };

// ---------------------------------------------------------------------------
// Kernel 1: per-code prep. One block (256 thr) per codebook row j.
// ---------------------------------------------------------------------------
__global__ __launch_bounds__(256, 1)
void vq_prep(const float* __restrict__ weight, _Float16* __restrict__ emb,
             float* __restrict__ cnorm, float* __restrict__ scl,
             float* __restrict__ diffAcc) {
  __shared__ float sred[256];
  const int j   = blockIdx.x;
  const int tid = threadIdx.x;
  const float* wr = weight + (size_t)j * DIM;
  float a0 = wr[tid];
  float a1 = wr[tid + 256];
  sred[tid] = a0 * a0 + a1 * a1;
  __syncthreads();
  for (int s = 128; s > 0; s >>= 1) {
    if (tid < s) sred[tid] += sred[tid + s];
    __syncthreads();
  }
  float norm2 = sred[0];
  float inv   = 1.0f / sqrtf(norm2);
  emb[(size_t)j * DIM + tid]       = (_Float16)(a0 * inv);
  emb[(size_t)j * DIM + tid + 256] = (_Float16)(a1 * inv);
  if (tid == 0) {
    cnorm[j] = norm2 * inv * inv;
    float n  = sqrtf(norm2);
    scl[j]   = fminf(1.0f, 1.0f / fmaxf(n, 1e-7f));
    if (j == 0) *diffAcc = 0.0f;
  }
}

// ---------------------------------------------------------------------------
// Kernel 2: main. 512 blocks x 64 threads (2 waves). Each block owns 64 rows;
// each wave owns 32 rows = 2 M-tiles, register-blocked 2M x 4N:
//   - 8 independent WMMA accumulator chains (no RAW stalls)
//   - each B fragment feeds 2 WMMAs -> codebook L2 traffic halved to ~1 GB
//   - A fragments from LDS once per K-step per M-tile
// ---------------------------------------------------------------------------
__global__ __launch_bounds__(64, 1)
void vq_main(const float* __restrict__ x, const float* __restrict__ weight,
             const _Float16* __restrict__ emb, const float* __restrict__ cnorm,
             const float* __restrict__ scl, float* __restrict__ out,
             float* __restrict__ diffAcc) {
  __shared__ __align__(16) unsigned char smem[65536];   // 64KB x-tile f16, reused later
  _Float16* xT = (_Float16*)smem;

  const int tid  = threadIdx.x;
  const int row0 = blockIdx.x * 64;

  // ---- stage 64 rows of x into LDS as f16 ----
  const float4* xv = (const float4*)(x + (size_t)row0 * DIM);
  #pragma unroll 4
  for (int i = tid; i < 64 * (DIM / 4); i += 64) {
    float4 f = xv[i];
    v4h h;
    h[0] = (_Float16)f.x; h[1] = (_Float16)f.y;
    h[2] = (_Float16)f.z; h[3] = (_Float16)f.w;
    *(v4h*)(xT + (size_t)i * 4) = h;
  }
  __syncthreads();

  const int l  = tid & 31;        // lane in wave32
  const int w  = tid >> 5;        // wave id (0..1), owns rows w*32..w*32+31
  const int g  = l >> 4;          // lane half-group
  const int mn = l & 15;          // A: row-in-tile / B: col-in-tile

  const _Float16* arow0 = xT + (size_t)(w * 32 + mn) * DIM;   // M-tile 0
  const _Float16* arow1 = arow0 + (size_t)16 * DIM;           // M-tile 1
  const _Float16* brow0 = emb + (size_t)mn * DIM;             // B fragment col

  float best[2][8];
  int   bidx[2][8];
  #pragma unroll
  for (int m = 0; m < 2; ++m)
    #pragma unroll
    for (int r = 0; r < 8; ++r) { best[m][r] = 3.402823466e38f; bidx[m][r] = 0; }

  for (int t4 = 0; t4 < 16; ++t4) {            // 16 groups of 4 N-tiles (64 codes)
    v8f acc[2][4];
    #pragma unroll
    for (int m = 0; m < 2; ++m)
      #pragma unroll
      for (int j = 0; j < 4; ++j)
        acc[m][j] = (v8f){0.f, 0.f, 0.f, 0.f, 0.f, 0.f, 0.f, 0.f};
    const _Float16* bbase = brow0 + (size_t)t4 * 64 * DIM;
    #pragma unroll 2
    for (int k = 0; k < 16; ++k) {             // K = 512 in steps of 32
      const int k0 = k * 32;
      H16 a0, a1;
      a0.h[0] = *(const v8h*)(arow0 + k0 + g * 8);
      a0.h[1] = *(const v8h*)(arow0 + k0 + 16 + g * 8);
      a1.h[0] = *(const v8h*)(arow1 + k0 + g * 8);
      a1.h[1] = *(const v8h*)(arow1 + k0 + 16 + g * 8);
      #pragma unroll
      for (int j = 0; j < 4; ++j) {
        H16 b;
        const _Float16* brow = bbase + (size_t)j * 16 * DIM;
        b.h[0] = *(const v8h*)(brow + k0 + g * 8);
        b.h[1] = *(const v8h*)(brow + k0 + 16 + g * 8);
        acc[0][j] = __builtin_amdgcn_wmma_f32_16x16x32_f16(
            false, a0.v, false, b.v, (short)0, acc[0][j], false, false);
        acc[1][j] = __builtin_amdgcn_wmma_f32_16x16x32_f16(
            false, a1.v, false, b.v, (short)0, acc[1][j], false, false);
      }
    }
    #pragma unroll
    for (int j = 0; j < 4; ++j) {
      const int   cand = (t4 * 4 + j) * 16 + mn;
      const float c    = cnorm[cand];
      #pragma unroll
      for (int m = 0; m < 2; ++m)
        #pragma unroll
        for (int r = 0; r < 8; ++r) {          // C elem (M=r+8g, N=mn): dist = c - 2*dot
          float s = fmaf(-2.0f, acc[m][j][r], c);
          if (s < best[m][r]) { best[m][r] = s; bidx[m][r] = cand; }
        }
    }
  }

  // ---- argmin across the 16 N-lanes (xor bits 0..3); first-index tiebreak ----
  for (int off = 1; off < 16; off <<= 1) {
    #pragma unroll
    for (int m = 0; m < 2; ++m)
      #pragma unroll
      for (int r = 0; r < 8; ++r) {
        float s2 = __shfl_xor(best[m][r], off, 32);
        int   i2 = __shfl_xor(bidx[m][r], off, 32);
        if (s2 < best[m][r] || (s2 == best[m][r] && i2 < bidx[m][r])) {
          best[m][r] = s2; bidx[m][r] = i2;
        }
      }
  }

  __syncthreads();                    // x-tile no longer needed; reuse LDS
  int* idxT = (int*)smem;
  if (mn == 0) {                      // lanes 0 and 16 hold rows r+8g per M-tile
    #pragma unroll
    for (int m = 0; m < 2; ++m)
      #pragma unroll
      for (int r = 0; r < 8; ++r)
        idxT[w * 32 + m * 16 + g * 8 + r] = bidx[m][r];
  }
  __syncthreads();

  // ---- phase 2: gather renormed rows, write output, accumulate MSE ----
  float lsum = 0.0f;
  const float4* wv4  = (const float4*)weight;
  const float4* x4   = (const float4*)x;
  float4*       out4 = (float4*)out;
  for (int i = tid; i < 64 * (DIM / 4); i += 64) {
    int row = i >> 7;
    int c4  = i & 127;
    int idx = idxT[row];
    float  s = scl[idx];
    float4 q = wv4[(size_t)idx * (DIM / 4) + c4];
    q.x *= s; q.y *= s; q.z *= s; q.w *= s;
    size_t go = (size_t)(row0 + row) * (DIM / 4) + c4;
    float4 xf = x4[go];
    float dx = q.x - xf.x, dy = q.y - xf.y, dz = q.z - xf.z, dw = q.w - xf.w;
    lsum += dx * dx + dy * dy + dz * dz + dw * dw;
    out4[go] = q;                    // (quantize + straight-through)/2 == quantize
  }
  #pragma unroll
  for (int off = 16; off > 0; off >>= 1) lsum += __shfl_xor(lsum, off, 32);
  float* red = (float*)(smem + 4096);          // clear of idxT region
  if (l == 0) red[w] = lsum;
  __syncthreads();
  if (tid == 0) atomicAdd(diffAcc, red[0] + red[1]);

  if (tid < 64) out[QN + 1 + row0 + tid] = (float)idxT[tid];  // indices as floats
}

// ---------------------------------------------------------------------------
// Kernel 3: finalize scalar diff
// ---------------------------------------------------------------------------
__global__ void vq_fin(const float* __restrict__ diffAcc, float* __restrict__ out) {
  out[QN] = *diffAcc * (1.0f / 16777216.0f);
}

extern "C" void kernel_launch(void* const* d_in, const int* in_sizes, int n_in,
                              void* d_out, int out_size, void* d_ws, size_t ws_size,
                              hipStream_t stream) {
  const float* x      = (const float*)d_in[0];   // [64,512,512] f32
  const float* weight = (const float*)d_in[1];   // [1024,512]   f32
  float* out = (float*)d_out;

  char* ws = (char*)d_ws;
  _Float16* emb   = (_Float16*)ws;                     // 1024*512*2 = 1 MiB
  float*    cnorm = (float*)(ws + (size_t)1048576);    // 4 KiB
  float*    scl   = cnorm + NEMB;                      // 4 KiB
  float*    dAcc  = scl + NEMB;                        // 4 B

  vq_prep<<<NEMB, 256, 0, stream>>>(weight, emb, cnorm, scl, dAcc);
  vq_main<<<NROWS / 64, 64, 0, stream>>>(x, weight, emb, cnorm, scl, out, dAcc);
  vq_fin<<<1, 1, 0, stream>>>(dAcc, out);
}
